// GCN_2800318677433
// MI455X (gfx1250) — matrix operand order repaired and checked by reading, not compiled, not run on hardware
//
#include <hip/hip_runtime.h>

typedef __attribute__((ext_vector_type(16))) _Float16 v16h;
typedef __attribute__((ext_vector_type(8)))  _Float16 v8h;
typedef __attribute__((ext_vector_type(8)))  float    v8f;

#define K_FEATS 128   // in_feats for both layers (layer2 input is 128-wide h1)
#define MT 4          // M-tiles (of 16 rows) per block

// ---------------------------------------------------------------- utilities
__global__ void zero_f32(float* __restrict__ p, long long n) {
  long long i = (long long)blockIdx.x * blockDim.x + threadIdx.x;
  if (i < n) p[i] = 0.0f;
}

__global__ void degree_kernel(const int* __restrict__ src, const int* __restrict__ dst,
                              float* __restrict__ degS, float* __restrict__ degD, int nE) {
  int e = blockIdx.x * blockDim.x + threadIdx.x;
  if (e < nE) {
    atomicAdd(&degS[src[e]], 1.0f);
    atomicAdd(&degD[dst[e]], 1.0f);
  }
}

__global__ void rsqrt_kernel(float* __restrict__ d, int n) {
  int i = blockIdx.x * blockDim.x + threadIdx.x;
  if (i < n) d[i] = rsqrtf(fmaxf(d[i], 1.0f));
}

// ---------------------------------------------------------------- WMMA GEMM
// H[M x N] = (X[M x 128] * norm[M]) @ W[128 x N], fp32 in, f16 multiply, f32 acc.
// N is a template constant so all addressing is shifts + immediate offsets.
// Block: 32*(N/16) threads; wave w computes cols [16w,16w+16) for MT row-tiles.
// W is staged TRANSPOSED in LDS so each lane's B fragment is 16 contiguous halfs.
template <int N>
__global__ void gemm_norm_wmma(const float* __restrict__ X, const float* __restrict__ W,
                               const float* __restrict__ norm, float* __restrict__ H,
                               int M) {
  __shared__ __align__(32) _Float16 ldsA[MT * 16 * K_FEATS];   // 16 KB A tiles
  __shared__ __align__(32) _Float16 ldsWt[N * K_FEATS];        // W^T: [n][k]

  const int tid     = threadIdx.x;
  const int lane    = tid & 31;
  const int wave    = tid >> 5;
  const int rowBase = blockIdx.x * (MT * 16);

  // stage W transposed: ldsWt[n*K + k] = W[k*N + n]   (N, K compile-time)
  for (int i = tid; i < K_FEATS * N; i += blockDim.x) {
    int k = i / N, n = i % N;
    ldsWt[n * K_FEATS + k] = (_Float16)W[i];
  }
  // stage A tiles, scaled by norm_src, zero-padded past M
  for (int i = tid; i < MT * 16 * K_FEATS; i += blockDim.x) {
    int r = i >> 7, c = i & 127;
    int row = rowBase + r;
    float v = 0.0f;
    if (row < M) v = X[(long long)row * K_FEATS + c] * norm[row];
    ldsA[i] = (_Float16)v;
  }
  __syncthreads();

  const int m    = lane & 15;          // A row within tile (ISA 16-bit A layout)
  const int ka   = (lane >> 4) * 16;   // A K-offset
  const int kb   = (lane >> 4) * 16;   // B K-offset (dense B layout)
  const int ncol = wave * 16 + (lane & 15);

  // Preload all four B fragments (whole K=128 for this lane's column):
  // fragment kt covers K = kt*32 + kb .. +15, contiguous in ldsWt.
  v16h bfrag[K_FEATS / 32];
  #pragma unroll
  for (int kt = 0; kt < K_FEATS / 32; ++kt)
    bfrag[kt] = *(const v16h*)&ldsWt[ncol * K_FEATS + kt * 32 + kb];

  #pragma unroll
  for (int mt = 0; mt < MT; ++mt) {
    const _Float16* ap = &ldsA[(mt * 16 + m) * K_FEATS];
    v8f acc = {};
    #pragma unroll
    for (int kt = 0; kt < K_FEATS / 32; ++kt) {
      const int k0 = kt * 32;
      // A: VGPRs 0-3 hold K = k0+ka .. +7 ; VGPRs 4-7 hold K = k0+8+ka .. +7
      v8h a0 = *(const v8h*)(ap + k0 + ka);
      v8h a1 = *(const v8h*)(ap + k0 + 8 + ka);
      v16h a = __builtin_shufflevector(a0, a1, 0, 1, 2, 3, 4, 5, 6, 7,
                                       8, 9, 10, 11, 12, 13, 14, 15);
      acc = __builtin_amdgcn_wmma_f32_16x16x32_f16(
          /*neg_a=*/false, a, /*neg_b=*/false, bfrag[kt],
          /*c_mod=*/(short)0, acc, /*reuse_a=*/false, /*reuse_b=*/false);
    }
    // C/D layout: VGPR j -> row = row0 + j, col = ncol
    const int row0 = rowBase + mt * 16 + (lane >> 4) * 8;
    float* hp = H + (size_t)row0 * N + ncol;
    if (row0 + 7 < M) {               // all full blocks take this path
      #pragma unroll
      for (int j = 0; j < 8; ++j) hp[j * N] = acc[j];   // immediate offsets
    } else {
      #pragma unroll
      for (int j = 0; j < 8; ++j)
        if (row0 + j < M) hp[j * N] = acc[j];
    }
  }
}

// ---------------------------------------------------------------- SpMM scatter
// AGG[dst[e]][:] += H[src[e]][:]   (one thread per (edge, 4 feats))
__global__ void spmm_scatter(const float* __restrict__ H, const int* __restrict__ src,
                             const int* __restrict__ dst, float* __restrict__ AGG,
                             int nE, int F) {
  const int f4 = F >> 2;
  long long t = (long long)blockIdx.x * blockDim.x + threadIdx.x;
  if (t >= (long long)nE * f4) return;
  int e = (int)(t / f4);
  int c = (int)(t % f4) * 4;
  const float4 v = *(const float4*)(H + (long long)src[e] * F + c);
  float* o = AGG + (long long)dst[e] * F + c;
  atomicAdd(o + 0, v.x);
  atomicAdd(o + 1, v.y);
  atomicAdd(o + 2, v.z);
  atomicAdd(o + 3, v.w);
}

// out = relu(agg * norm_dst + bias), in place
__global__ void finalize_kernel(float* __restrict__ AGG, const float* __restrict__ normD,
                                const float* __restrict__ bias, int M, int F) {
  long long t = (long long)blockIdx.x * blockDim.x + threadIdx.x;
  if (t >= (long long)M * F) return;
  int row = (int)(t / F), c = (int)(t % F);
  float v = AGG[t] * normD[row] + bias[c];
  AGG[t] = fmaxf(v, 0.0f);
}

// ---------------------------------------------------------------- mean pooling
__global__ void pool_accum(const float* __restrict__ H, const int* __restrict__ gid,
                           float* __restrict__ sums, float* __restrict__ counts,
                           int M, int F) {
  long long t = (long long)blockIdx.x * blockDim.x + threadIdx.x;
  if (t >= (long long)M * F) return;
  int node = (int)(t / F), c = (int)(t % F);
  int g = gid[node];
  atomicAdd(&sums[(long long)g * F + c], H[t]);
  if (c == 0) atomicAdd(&counts[g], 1.0f);
}

__global__ void pool_div(const float* __restrict__ sums, const float* __restrict__ counts,
                         float* __restrict__ out, int nG, int F) {
  int t = blockIdx.x * blockDim.x + threadIdx.x;
  if (t < nG * F) out[t] = sums[t] / fmaxf(counts[t / F], 1.0f);
}

// ---------------------------------------------------------------- launch
extern "C" void kernel_launch(void* const* d_in, const int* in_sizes, int n_in,
                              void* d_out, int out_size, void* d_ws, size_t ws_size,
                              hipStream_t stream) {
  (void)n_in; (void)ws_size;
  const float* features = (const float*)d_in[0];
  const float* W1       = (const float*)d_in[1];
  const float* b1       = (const float*)d_in[2];
  const float* W2       = (const float*)d_in[3];
  const float* b2       = (const float*)d_in[4];
  const int*   src      = (const int*)d_in[5];
  const int*   dst      = (const int*)d_in[6];
  const int*   gids     = (const int*)d_in[7];

  const int M  = in_sizes[7];            // n_nodes
  const int nE = in_sizes[5];            // n_edges
  const int F1 = in_sizes[2];            // in/hidden feats (128)
  const int F2 = in_sizes[4];            // out feats (64)
  const int nG = out_size / F2;          // num_graphs

  char* ws = (char*)d_ws;
  size_t off = 0;
  auto alloc = [&](size_t bytes) -> float* {
    float* p = (float*)(ws + off);
    off = (off + bytes + 255) & ~(size_t)255;
    return p;
  };
  float* normS = alloc((size_t)M * sizeof(float));
  float* normD = alloc((size_t)M * sizeof(float));
  float* bufH  = alloc((size_t)M * F1 * sizeof(float));   // GEMM outputs
  float* bufA  = alloc((size_t)M * F1 * sizeof(float));   // aggregation / activations
  float* sums   = bufH;                 // reused after layer-2 GEMM is consumed
  float* counts = bufH + (size_t)nG * F2;

  const int T = 256;
  auto blocks = [](long long n, int t) { return (unsigned)((n + t - 1) / t); };

  // degrees -> rsqrt norms
  zero_f32<<<blocks(M, T), T, 0, stream>>>(normS, M);
  zero_f32<<<blocks(M, T), T, 0, stream>>>(normD, M);
  degree_kernel<<<blocks(nE, T), T, 0, stream>>>(src, dst, normS, normD, nE);
  rsqrt_kernel<<<blocks(M, T), T, 0, stream>>>(normS, M);
  rsqrt_kernel<<<blocks(M, T), T, 0, stream>>>(normD, M);

  const unsigned gemmGrid = (unsigned)((M + MT * 16 - 1) / (MT * 16));

  // ---- layer 1: H1 = (X * normS) @ W1 ; agg ; relu(agg*normD + b1)
  gemm_norm_wmma<128><<<gemmGrid, 32 * (128 / 16), 0, stream>>>(features, W1, normS, bufH, M);
  zero_f32<<<blocks((long long)M * F1, T), T, 0, stream>>>(bufA, (long long)M * F1);
  spmm_scatter<<<blocks((long long)nE * (F1 / 4), T), T, 0, stream>>>(bufH, src, dst, bufA, nE, F1);
  finalize_kernel<<<blocks((long long)M * F1, T), T, 0, stream>>>(bufA, normD, b1, M, F1);

  // ---- layer 2: H2 = (h1 * normS) @ W2 ; agg ; relu(agg*normD + b2)
  gemm_norm_wmma<64><<<gemmGrid, 32 * (64 / 16), 0, stream>>>(bufA, W2, normS, bufH, M);
  zero_f32<<<blocks((long long)M * F2, T), T, 0, stream>>>(bufA, (long long)M * F2);
  spmm_scatter<<<blocks((long long)nE * (F2 / 4), T), T, 0, stream>>>(bufH, src, dst, bufA, nE, F2);
  finalize_kernel<<<blocks((long long)M * F2, T), T, 0, stream>>>(bufA, normD, b2, M, F2);

  // ---- per-graph mean pooling -> d_out
  zero_f32<<<blocks((long long)nG * F2 + nG, T), T, 0, stream>>>(sums, (long long)nG * F2 + nG);
  pool_accum<<<blocks((long long)M * F2, T), T, 0, stream>>>(bufA, gids, sums, counts, M, F2);
  pool_div<<<blocks((long long)nG * F2, T), T, 0, stream>>>(sums, counts, (float*)d_out, nG, F2);
}
